// Self_Attn_66726611910773
// MI455X (gfx1250) — matrix operand order repaired and checked by reading, not compiled, not run on hardware
//
#include <hip/hip_runtime.h>
#include <stdint.h>

#define B_  4
#define C_  256
#define N_  4096
#define OC_ 32

typedef __attribute__((ext_vector_type(16))) _Float16 v16h;
typedef __attribute__((ext_vector_type(8)))  _Float16 v8h;
typedef __attribute__((ext_vector_type(8)))  float    v8f;

union V16U { v16h v; v8h p[2]; };
union V16W { v16h v; uint32_t u[8]; };
union H2U  { uint32_t u; _Float16 h[2]; };

__device__ __forceinline__ v16h combine16(v8h lo, v8h hi) {
  V16U t; t.p[0] = lo; t.p[1] = hi; return t.v;
}
__device__ __forceinline__ uint32_t pack2(float a, float b) {
  H2U t; t.h[0] = (_Float16)a; t.h[1] = (_Float16)b; return t.u;
}
__device__ __forceinline__ v8f zero8() {
  v8f z;
#pragma unroll
  for (int r = 0; r < 8; ++r) z[r] = 0.f;
  return z;
}
__device__ __forceinline__ v8f wmma16(v16h a, v16h b, v8f c) {
  // D = A(16x32 f16) x B(32x16 f16) + C(16x16 f32)
  return __builtin_amdgcn_wmma_f32_16x16x32_f16(false, a, false, b, (short)0, c,
                                                false, false);
}

// ---------------------------------------------------------------------------
// Kernel 1: transpose+convert x[B][C][N] f32 -> Xt[B][N][C] f16 (LDS tiles)
// ---------------------------------------------------------------------------
__global__ void k_transpose(const float* __restrict__ x, _Float16* __restrict__ Xt) {
  __shared__ float tile[32][33];
  int b  = blockIdx.z;
  int n0 = blockIdx.x * 32;
  int c0 = blockIdx.y * 32;
  int tx = threadIdx.x;  // 0..31
  int ty = threadIdx.y;  // 0..7
#pragma unroll
  for (int i = 0; i < 32; i += 8)
    tile[ty + i][tx] = x[((size_t)(b * C_ + c0 + ty + i)) * N_ + n0 + tx];
  __syncthreads();
#pragma unroll
  for (int i = 0; i < 32; i += 8)
    Xt[((size_t)(b * N_ + n0 + ty + i)) * C_ + c0 + tx] = (_Float16)tile[tx][ty + i];
}

// ---------------------------------------------------------------------------
// Kernel 2: convert weights to f16 (row-major [OC][C])
// ---------------------------------------------------------------------------
__global__ void k_cvt_w(const float* __restrict__ Wq, const float* __restrict__ Wk,
                        const float* __restrict__ Wv, _Float16* __restrict__ Wqh,
                        _Float16* __restrict__ Wkh, _Float16* __restrict__ Wvh) {
  int i = blockIdx.x * blockDim.x + threadIdx.x;
  if (i < OC_ * C_) {
    Wqh[i] = (_Float16)Wq[i];
    Wkh[i] = (_Float16)Wk[i];
  }
  if (i < C_ * C_) Wvh[i] = (_Float16)Wv[i];
}

// ---------------------------------------------------------------------------
// Kernel 3: projections via WMMA.
//   Qh[b][n][oc] = sum_c Xt[b][n][c]*Wq[oc][c] + bq[oc]   (same for K)
//   Vh[b][oc][n] = sum_c Wv[oc][c]*Xt[b][n][c] + bv[oc]   (channel-major)
// One wave per 16-query tile. 1024 tiles total.
// ---------------------------------------------------------------------------
__global__ void __launch_bounds__(256)
k_proj(const _Float16* __restrict__ Xt,
       const _Float16* __restrict__ Wqh, const float* __restrict__ bq,
       const _Float16* __restrict__ Wkh, const float* __restrict__ bk,
       const _Float16* __restrict__ Wvh, const float* __restrict__ bv,
       _Float16* __restrict__ Qh, _Float16* __restrict__ Kh,
       _Float16* __restrict__ Vh) {
  int lane = threadIdx.x & 31;
  int wid  = blockIdx.x * (blockDim.x >> 5) + (threadIdx.x >> 5);
  int b  = wid >> 8;           // 256 query tiles per batch
  int q0 = (wid & 255) << 4;
  int hs = lane >> 4;          // half-wave select
  int lr = lane & 15;

  const _Float16* Xb = Xt + (size_t)b * N_ * C_;

  // ---- Q,K tiles: D[m=query][n=oc], 2 oc-tiles each ----
  for (int t = 0; t < 4; ++t) {
    const _Float16* W = (t < 2) ? Wqh : Wkh;
    int oc0 = (t & 1) * 16;
    v8f acc = zero8();
#pragma unroll
    for (int c0 = 0; c0 < C_; c0 += 32) {
      // A = Xt rows [16 q x 32 ch]
      const _Float16* pa = Xb + (size_t)(q0 + lr) * C_ + c0 + hs * 16;
      v16h A = combine16(*(const v8h*)pa, *(const v8h*)(pa + 8));
      // B = W^T [32 ch x 16 oc]: lane = oc column, contiguous channels
      const _Float16* pb = W + (size_t)(oc0 + lr) * C_ + c0 + hs * 16;
      v16h Bm = combine16(*(const v8h*)pb, *(const v8h*)(pb + 8));
      acc = wmma16(A, Bm, acc);
    }
    const float* bias = (t < 2) ? bq : bk;
    _Float16* Out = (t < 2) ? Qh : Kh;
    float bsc = bias[oc0 + lr];
#pragma unroll
    for (int r = 0; r < 8; ++r) {
      int q = q0 + r + hs * 8;
      Out[((size_t)(b * N_) + q) * OC_ + oc0 + lr] = (_Float16)(acc[r] + bsc);
    }
  }

  // ---- V tiles: D[m=oc][n=query], 16 oc-tiles ----
  for (int t = 0; t < 16; ++t) {
    int oc0 = t * 16;
    v8f acc = zero8();
#pragma unroll
    for (int c0 = 0; c0 < C_; c0 += 32) {
      // A = Wv rows [16 oc x 32 ch]
      const _Float16* pa = Wvh + (size_t)(oc0 + lr) * C_ + c0 + hs * 16;
      v16h A = combine16(*(const v8h*)pa, *(const v8h*)(pa + 8));
      // B = Xt^T [32 ch x 16 q]: lane = query column, contiguous channels
      const _Float16* pb = Xb + (size_t)(q0 + lr) * C_ + c0 + hs * 16;
      v16h Bm = combine16(*(const v8h*)pb, *(const v8h*)(pb + 8));
      acc = wmma16(A, Bm, acc);
    }
#pragma unroll
    for (int r = 0; r < 8; ++r) {
      int oc = oc0 + r + hs * 8;
      Vh[((size_t)(b * C_) + oc) * N_ + q0 + lr] = (_Float16)(acc[r] + bv[oc]);
    }
  }
}

// ---------------------------------------------------------------------------
// Kernel 4: fused flash attention + residual.
// Per wave: one 16-query tile, all 256 output channels (16 acc tiles).
// E^T = K_rows x Q_cols (K-dim = 32 channels, exactly one WMMA per 16 keys).
// Softmax over keys = per-lane VGPR reduction + one shfl_xor(16).
// P -> B-operand for PV via pack + one cross-half shuffle (no LDS).
// ---------------------------------------------------------------------------
__global__ void __launch_bounds__(256)
k_attn(const _Float16* __restrict__ Qh, const _Float16* __restrict__ Kh,
       const _Float16* __restrict__ Vh, const float* __restrict__ x,
       const float* __restrict__ gamma, float* __restrict__ out) {
  int lane = threadIdx.x & 31;
  int wid  = blockIdx.x * (blockDim.x >> 5) + (threadIdx.x >> 5);
  int b  = wid >> 8;
  int i0 = (wid & 255) << 4;
  int hs = lane >> 4;
  int lr = lane & 15;

  // Q operand (B matrix [32 ch x 16 queries]) loaded once, reused for all keys
  const _Float16* pq = Qh + ((size_t)(b * N_) + i0 + lr) * OC_ + hs * 16;
  v16h Bq = combine16(*(const v8h*)pq, *(const v8h*)(pq + 8));

  v8f acc[16];
#pragma unroll
  for (int t = 0; t < 16; ++t) acc[t] = zero8();
  float mrun = -__builtin_inff();
  float lrun = 0.f;

  const _Float16* Kb = Kh + (size_t)(b * N_) * OC_;
  const _Float16* Vb = Vh + (size_t)(b * C_) * N_;

  for (int j0 = 0; j0 < N_; j0 += 32) {
    // E^T tiles: rows = keys, cols = queries
    const _Float16* pk0 = Kb + (size_t)(j0 + lr) * OC_ + hs * 16;
    v16h A0 = combine16(*(const v8h*)pk0, *(const v8h*)(pk0 + 8));
    const _Float16* pk1 = Kb + (size_t)(j0 + 16 + lr) * OC_ + hs * 16;
    v16h A1 = combine16(*(const v8h*)pk1, *(const v8h*)(pk1 + 8));
    v8f e0 = wmma16(A0, Bq, zero8());
    v8f e1 = wmma16(A1, Bq, zero8());

    // block max over 32 keys for this query column
    float bm = e0[0];
#pragma unroll
    for (int r = 1; r < 8; ++r) bm = fmaxf(bm, e0[r]);
#pragma unroll
    for (int r = 0; r < 8; ++r) bm = fmaxf(bm, e1[r]);
    bm = fmaxf(bm, __shfl_xor(bm, 16, 32));

    float mn    = fmaxf(mrun, bm);
    float scale = __expf(mrun - mn);   // first iter: exp(-inf) = 0
    float p0[8], p1[8];
    float bs = 0.f;
#pragma unroll
    for (int r = 0; r < 8; ++r) { p0[r] = __expf(e0[r] - mn); bs += p0[r]; }
#pragma unroll
    for (int r = 0; r < 8; ++r) { p1[r] = __expf(e1[r] - mn); bs += p1[r]; }
    bs += __shfl_xor(bs, 16, 32);
    lrun = lrun * scale + bs;
    mrun = mn;

#pragma unroll
    for (int t = 0; t < 16; ++t) {
#pragma unroll
      for (int r = 0; r < 8; ++r) acc[t][r] *= scale;
    }

    // Build P operand (B matrix [32 keys x 16 queries]) in-register:
    // lanes 0-15 need keys j0+0..15  = {own tile0, partner tile0}
    // lanes 16-31 need keys j0+16..31 = {partner tile1, own tile1}
    uint32_t u0[4], u1[4];
#pragma unroll
    for (int r = 0; r < 4; ++r) {
      u0[r] = pack2(p0[2 * r], p0[2 * r + 1]);
      u1[r] = pack2(p1[2 * r], p1[2 * r + 1]);
    }
    V16W bw;
#pragma unroll
    for (int r = 0; r < 4; ++r) {
      uint32_t o0 = __shfl_xor(u0[r], 16, 32);
      uint32_t o1 = __shfl_xor(u1[r], 16, 32);
      bw.u[r]     = hs ? o1 : u0[r];
      bw.u[4 + r] = hs ? u1[r] : o0;
    }
    v16h B2 = bw.v;

    // acc[t] += V_tile(16 ch x 32 keys) x P(32 keys x 16 queries)
#pragma unroll
    for (int t = 0; t < 16; ++t) {
      const _Float16* pv = Vb + (size_t)(t * 16 + lr) * N_ + j0 + hs * 16;
      v16h Av = combine16(*(const v8h*)pv, *(const v8h*)(pv + 8));
      acc[t] = wmma16(Av, B2, acc[t]);
    }
  }

  float inv = 1.f / lrun;
  float g   = gamma[0];
  size_t xbase = (size_t)b * C_ * N_ + i0 + lr;
#pragma unroll
  for (int t = 0; t < 16; ++t) {
#pragma unroll
    for (int r = 0; r < 8; ++r) {
      int c = t * 16 + r + hs * 8;
      size_t idx = xbase + (size_t)c * N_;
      out[idx] = g * (acc[t][r] * inv) + x[idx];
    }
  }
}

// ---------------------------------------------------------------------------
extern "C" void kernel_launch(void* const* d_in, const int* in_sizes, int n_in,
                              void* d_out, int out_size, void* d_ws, size_t ws_size,
                              hipStream_t stream) {
  (void)in_sizes; (void)n_in; (void)out_size;
  const float* x     = (const float*)d_in[0];
  const float* Wq    = (const float*)d_in[1];
  const float* bq    = (const float*)d_in[2];
  const float* Wk    = (const float*)d_in[3];
  const float* bk    = (const float*)d_in[4];
  const float* Wv    = (const float*)d_in[5];
  const float* bv    = (const float*)d_in[6];
  const float* gamma = (const float*)d_in[7];
  float* out = (float*)d_out;

  char* ws = (char*)d_ws;
  size_t o = 0;
  _Float16* Xt  = (_Float16*)(ws + o); o += (size_t)B_ * N_ * C_ * 2;   // 8 MB
  _Float16* Qh  = (_Float16*)(ws + o); o += (size_t)B_ * N_ * OC_ * 2;  // 1 MB
  _Float16* Kh  = (_Float16*)(ws + o); o += (size_t)B_ * N_ * OC_ * 2;  // 1 MB
  _Float16* Vh  = (_Float16*)(ws + o); o += (size_t)B_ * C_ * N_ * 2;   // 8 MB
  _Float16* Wqh = (_Float16*)(ws + o); o += (size_t)OC_ * C_ * 2;
  _Float16* Wkh = (_Float16*)(ws + o); o += (size_t)OC_ * C_ * 2;
  _Float16* Wvh = (_Float16*)(ws + o); o += (size_t)C_ * C_ * 2;
  if (ws_size < o) return;  // ~18.2 MB scratch needed

  k_transpose<<<dim3(N_ / 32, C_ / 32, B_), dim3(32, 8), 0, stream>>>(x, Xt);
  k_cvt_w<<<(C_ * C_ + 255) / 256, 256, 0, stream>>>(Wq, Wk, Wv, Wqh, Wkh, Wvh);
  int waves = B_ * (N_ / 16);  // 1024 independent 16-query tiles
  k_proj<<<waves / 8, 256, 0, stream>>>(Xt, Wqh, bq, Wkh, bk, Wvh, bv, Qh, Kh, Vh);
  k_attn<<<waves / 8, 256, 0, stream>>>(Qh, Kh, Vh, x, gamma, out);
}
